// RelativeAttention_57810259804634
// MI455X (gfx1250) — compile-verified, matrix operand order checked by reading, not run on hardware
//
#include <hip/hip_runtime.h>

#define B_DIM 16384
#define A_DIM 4096
#define D_DIM 512
#define H_DIM 512

typedef __attribute__((ext_vector_type(16))) __bf16 v16bf;
typedef __attribute__((ext_vector_type(8)))  __bf16 v8bf;
typedef __attribute__((ext_vector_type(8)))  float  v8f;
typedef int ra_v4i __attribute__((vector_size(16)));

// ---------------------------------------------------------------------------
// CDNA5 async global->LDS copy (ASYNCcnt path), with synchronous fallback.
// Builtin prototype (from hipcc diagnostic): (v4i AS1*, v4i AS3*, Ii, Ii).
// ---------------------------------------------------------------------------
#if defined(__has_builtin)
#  if __has_builtin(__builtin_amdgcn_global_load_async_to_lds_b128)
#    define RA_ASYNC 1
#  endif
#endif
#ifndef RA_ASYNC
#  define RA_ASYNC 0
#endif

__device__ __forceinline__ void ra_cp16(__bf16* dst_lds, const __bf16* src_glb) {
#if RA_ASYNC
  __builtin_amdgcn_global_load_async_to_lds_b128(
      (__attribute__((address_space(1))) ra_v4i*)(ra_v4i*)src_glb,
      (__attribute__((address_space(3))) ra_v4i*)(ra_v4i*)dst_lds,
      /*offset=*/0, /*cpol=*/0);
#else
  *(v8bf*)dst_lds = *(const v8bf*)src_glb;
#endif
}

__device__ __forceinline__ void ra_wait_async() {
#if RA_ASYNC
#  if __has_builtin(__builtin_amdgcn_s_wait_asynccnt)
  __builtin_amdgcn_s_wait_asynccnt(0);
#  else
  asm volatile("s_wait_asynccnt 0" ::: "memory");
#  endif
#endif
}

// ---------------------------------------------------------------------------
// WMMA helper: D = A(16x32 bf16) * B(32x16 bf16) + C(16x16 f32)
// ---------------------------------------------------------------------------
__device__ __forceinline__ v8f wmma_bf16(v16bf a, v16bf b, v8f c) {
  return __builtin_amdgcn_wmma_f32_16x16x32_bf16(
      /*neg_a=*/false, a, /*neg_b=*/false, b,
      /*c_mod=*/(short)0, c, /*reuse_a=*/false, /*reuse_b=*/false);
}

// A/B fragment from a row-major bf16 matrix (works for LDS or global ptrs).
// 16-bit operand layout (ISA 7.12.2): lane&15 selects row (A) / col (B);
// lanes>=16 take K+8; per lane: 8 bf16 at K..K+7 and 8 bf16 at K+16..K+23.
__device__ __forceinline__ v16bf frag_bf16(const __bf16* base, int ld,
                                           int row0, int k0, int lane) {
  int r  = row0 + (lane & 15);
  int kb = k0 + ((lane & 16) ? 8 : 0);
  const __bf16* p = base + (size_t)r * ld + kb;
  v8bf lo = *(const v8bf*)(p);
  v8bf hi = *(const v8bf*)(p + 16);
  v16bf f;
#pragma unroll
  for (int i = 0; i < 8; ++i) { f[i] = lo[i]; f[i + 8] = hi[i]; }
  return f;
}

// Same fragment but sourced from f32 memory with on-the-fly bf16 convert.
__device__ __forceinline__ v16bf frag_f32(const float* base, int ld,
                                          int row0, int k0, int lane) {
  int r  = row0 + (lane & 15);
  int kb = k0 + ((lane & 16) ? 8 : 0);
  const float* p = base + (size_t)r * ld + kb;
  v16bf f;
#pragma unroll
  for (int i = 0; i < 8; ++i) { f[i] = (__bf16)p[i]; f[i + 8] = (__bf16)p[16 + i]; }
  return f;
}

// ---------------------------------------------------------------------------
// Kernel: zero the softmax row-sum accumulator
// ---------------------------------------------------------------------------
__global__ void ra_zero_sums(float* __restrict__ sums) {
  int i = blockIdx.x * blockDim.x + threadIdx.x;
  if (i < B_DIM) sums[i] = 0.0f;
}

// ---------------------------------------------------------------------------
// Kernel: values[A][H] f32  ->  VT[H][A] bf16   (coalesced writes, L2 reads)
// ---------------------------------------------------------------------------
__global__ void ra_transpose_values(const float* __restrict__ V,
                                    __bf16* __restrict__ VT) {
  int idx = blockIdx.x * blockDim.x + threadIdx.x;  // over H*A
  int h = idx >> 12;          // A_DIM == 4096 == 2^12
  int a = idx & (A_DIM - 1);
  VT[idx] = (__bf16)V[(size_t)a * H_DIM + h];
}

// ---------------------------------------------------------------------------
// Kernel: projection + L2 normalize.
// One workgroup (8 waves) computes 16 rows x 512 cols of X @ W^T, reduces the
// row norms through LDS, writes normalized bf16 rows.
// ---------------------------------------------------------------------------
__global__ __launch_bounds__(256) void ra_proj_norm(
    const float* __restrict__ X, const float* __restrict__ W,
    __bf16* __restrict__ OUT) {
  __shared__ __bf16 Atile[16 * 32];   // 1 KB : current K-slice of X (bf16)
  __shared__ float  Crow[16 * 512];   // 32 KB: full f32 result block
  __shared__ float  rinv[16];

  const int t = threadIdx.x;
  const int lane = t & 31;
  const int wave = t >> 5;
  const int row0 = blockIdx.x * 16;
  const int n0 = wave * 64;           // each wave owns 4 column subtiles

  v8f acc[4];
#pragma unroll
  for (int cs = 0; cs < 4; ++cs)
#pragma unroll
    for (int j = 0; j < 8; ++j) acc[cs][j] = 0.0f;

  for (int k0 = 0; k0 < D_DIM; k0 += 32) {
    {  // stage 16x32 slice of X as bf16 (2 elements / thread)
      int idx = t * 2;
      int r = idx >> 5, c = idx & 31;
      const float2 xv = *(const float2*)(X + (size_t)(row0 + r) * D_DIM + k0 + c);
      Atile[r * 32 + c]     = (__bf16)xv.x;
      Atile[r * 32 + c + 1] = (__bf16)xv.y;
    }
    __syncthreads();
    v16bf a = frag_bf16(Atile, 32, 0, 0, lane);
#pragma unroll
    for (int cs = 0; cs < 4; ++cs) {
      v16bf b = frag_f32(W, D_DIM, n0 + cs * 16, k0, lane);  // W row n = col n
      acc[cs] = wmma_bf16(a, b, acc[cs]);
    }
    __syncthreads();
  }

  // spill C to LDS (layout: VGPR r, lanes<16 -> M=r; lanes>=16 -> M=r+8)
#pragma unroll
  for (int cs = 0; cs < 4; ++cs)
#pragma unroll
    for (int r = 0; r < 8; ++r) {
      int row = r + ((lane & 16) ? 8 : 0);
      int col = n0 + cs * 16 + (lane & 15);
      Crow[row * 512 + col] = acc[cs][r];
    }
  __syncthreads();

  // row sum-of-squares: 16 threads per row, then shfl reduce within 16 lanes
  {
    int rr = t >> 4, c0 = t & 15;
    float s = 0.0f;
    for (int j = 0; j < 32; ++j) {
      float v = Crow[rr * 512 + c0 + j * 16];
      s += v * v;
    }
    for (int m = 8; m >= 1; m >>= 1) s += __shfl_xor(s, m, 16);
    if (c0 == 0) rinv[rr] = 1.0f / fmaxf(sqrtf(s), 1e-12f);
  }
  __syncthreads();

  // normalized bf16 output, coalesced
  for (int i = 0; i < 32; ++i) {
    int idx = t + i * 256;
    int row = idx >> 9, col = idx & 511;
    OUT[(size_t)(row0 + row) * H_DIM + col] = (__bf16)(Crow[idx] * rinv[row]);
  }
}

// ---------------------------------------------------------------------------
// Kernel: similarities + exp-sum.
// Workgroup computes a 64x128 tile of Q @ K^T (K-loop over H=512) with
// double-buffered async global->LDS staging; writes f32 sims and atomically
// accumulates per-row sum of exp(quantized).
// ---------------------------------------------------------------------------
__global__ __launch_bounds__(256) void ra_sim(
    const __bf16* __restrict__ Q, const __bf16* __restrict__ Kn,
    float* __restrict__ SIM, float* __restrict__ sums) {
  __shared__ __bf16 Qt[2][64 * 32];    // 2 x 4 KB
  __shared__ __bf16 Kt[2][128 * 32];   // 2 x 8 KB

  const int t = threadIdx.x;
  const int lane = t & 31;
  const int wave = t >> 5;
  const int wm = wave & 3;          // row subtile (16 rows)
  const int wn = wave >> 2;         // 64-col group (4 subtiles)
  const int rowblk = blockIdx.y * 64;
  const int colblk = blockIdx.x * 128;

  // per-thread staging coordinates (16B granules)
  const int qr = t >> 2, qc = (t & 3) * 8;          // Q: 64x32, 1 granule
  const int kr = t >> 1, kc = (t & 1) * 16;         // K: 128x32, 2 granules

  auto stage = [&](int k0, int buf) {
    ra_cp16(&Qt[buf][qr * 32 + qc],
            &Q[(size_t)(rowblk + qr) * H_DIM + k0 + qc]);
    const __bf16* src = &Kn[(size_t)(colblk + kr) * H_DIM + k0 + kc];
    ra_cp16(&Kt[buf][kr * 32 + kc],     src);
    ra_cp16(&Kt[buf][kr * 32 + kc + 8], src + 8);
  };

  v8f acc[4];
#pragma unroll
  for (int cs = 0; cs < 4; ++cs)
#pragma unroll
    for (int j = 0; j < 8; ++j) acc[cs][j] = 0.0f;

  stage(0, 0);
  ra_wait_async();
  __syncthreads();

  for (int k0 = 0; k0 < H_DIM; k0 += 32) {
    const int buf = (k0 >> 5) & 1;
    if (k0 + 32 < H_DIM) stage(k0 + 32, buf ^ 1);   // overlap copy with WMMA

    v16bf a = frag_bf16(Qt[buf], 32, wm * 16, 0, lane);
#pragma unroll
    for (int cs = 0; cs < 4; ++cs) {
      v16bf b = frag_bf16(Kt[buf], 32, wn * 64 + cs * 16, 0, lane);
      acc[cs] = wmma_bf16(a, b, acc[cs]);
    }
    ra_wait_async();
    __syncthreads();
  }

  // epilogue: write sims, accumulate exp(quantized) row partials
  float esum[8];
#pragma unroll
  for (int r = 0; r < 8; ++r) esum[r] = 0.0f;

#pragma unroll
  for (int cs = 0; cs < 4; ++cs)
#pragma unroll
    for (int r = 0; r < 8; ++r) {
      int row = rowblk + wm * 16 + r + ((lane & 16) ? 8 : 0);
      int col = colblk + wn * 64 + cs * 16 + (lane & 15);
      float s = acc[cs][r];
      SIM[(size_t)row * A_DIM + col] = s;
      float qz = rintf(s * 20.0f) * 0.05f;   // |s|<=1 -> exp is safe, no max
      esum[r] += __expf(qz);
    }

#pragma unroll
  for (int r = 0; r < 8; ++r) {
    float s = esum[r];
    for (int m = 8; m >= 1; m >>= 1) s += __shfl_xor(s, m, 16);
    if ((lane & 15) == 0) {
      int row = rowblk + wm * 16 + r + ((lane & 16) ? 8 : 0);
      atomicAdd(&sums[row], s);
    }
  }
}

// ---------------------------------------------------------------------------
// Kernel: output = softmax(quantized sims) @ values.
// Workgroup: 32 rows x full H=512. Re-reads sims from d_out, fuses
// quantize+exp+bf16 into LDS A-tile, WMMA against VT[H][A] (K-contiguous),
// divides by row sum in the epilogue.
// ---------------------------------------------------------------------------
__global__ __launch_bounds__(256) void ra_out(
    const float* __restrict__ SIM, const __bf16* __restrict__ VT,
    const float* __restrict__ sums, float* __restrict__ OUT) {
  __shared__ __bf16 Et[32 * 32];    // 2 KB

  const int t = threadIdx.x;
  const int lane = t & 31;
  const int wave = t >> 5;
  const int rowblk = blockIdx.x * 32;
  const int col0 = wave * 64;       // 4 column subtiles per wave

  v8f acc[2][4];
#pragma unroll
  for (int st = 0; st < 2; ++st)
#pragma unroll
    for (int cs = 0; cs < 4; ++cs)
#pragma unroll
      for (int j = 0; j < 8; ++j) acc[st][cs][j] = 0.0f;

  for (int k0 = 0; k0 < A_DIM; k0 += 32) {
    {  // stage exp(quantized sim) 32x32 as bf16 (float4 / thread)
      int r = t >> 3, c = (t & 7) * 4;
      const float4 s4 =
          *(const float4*)&SIM[(size_t)(rowblk + r) * A_DIM + k0 + c];
      Et[r * 32 + c + 0] = (__bf16)__expf(rintf(s4.x * 20.0f) * 0.05f);
      Et[r * 32 + c + 1] = (__bf16)__expf(rintf(s4.y * 20.0f) * 0.05f);
      Et[r * 32 + c + 2] = (__bf16)__expf(rintf(s4.z * 20.0f) * 0.05f);
      Et[r * 32 + c + 3] = (__bf16)__expf(rintf(s4.w * 20.0f) * 0.05f);
    }
    __syncthreads();
    v16bf a0 = frag_bf16(Et, 32,  0, 0, lane);
    v16bf a1 = frag_bf16(Et, 32, 16, 0, lane);
#pragma unroll
    for (int cs = 0; cs < 4; ++cs) {
      v16bf b = frag_bf16(VT, A_DIM, col0 + cs * 16, k0, lane);
      acc[0][cs] = wmma_bf16(a0, b, acc[0][cs]);
      acc[1][cs] = wmma_bf16(a1, b, acc[1][cs]);
    }
    __syncthreads();
  }

#pragma unroll
  for (int st = 0; st < 2; ++st)
#pragma unroll
    for (int r = 0; r < 8; ++r) {
      int row = rowblk + st * 16 + r + ((lane & 16) ? 8 : 0);
      float inv = 1.0f / sums[row];
#pragma unroll
      for (int cs = 0; cs < 4; ++cs) {
        int col = col0 + cs * 16 + (lane & 15);
        OUT[(size_t)row * H_DIM + col] = acc[st][cs][r] * inv;
      }
    }
}

// ---------------------------------------------------------------------------
// Host launcher
// ---------------------------------------------------------------------------
extern "C" void kernel_launch(void* const* d_in, const int* in_sizes, int n_in,
                              void* d_out, int out_size, void* d_ws, size_t ws_size,
                              hipStream_t stream) {
  (void)in_sizes; (void)n_in; (void)out_size; (void)ws_size;

  const float* x       = (const float*)d_in[0];  // [B, D]
  const float* anchors = (const float*)d_in[1];  // [A, D]
  const float* W_q     = (const float*)d_in[2];  // [H, D]
  const float* W_k     = (const float*)d_in[3];  // [H, D]
  const float* values  = (const float*)d_in[4];  // [A, H]

  float* out = (float*)d_out;                          // [B, H]
  float* sim = out + (size_t)B_DIM * H_DIM;            // [B, A]

  __bf16* q_bf = (__bf16*)d_ws;                        // [B, H] bf16
  __bf16* k_bf = q_bf + (size_t)B_DIM * H_DIM;         // [A, H] bf16
  __bf16* v_t  = k_bf + (size_t)A_DIM * H_DIM;         // [H, A] bf16
  float*  sums = (float*)(v_t + (size_t)A_DIM * H_DIM);// [B] f32

  ra_zero_sums<<<B_DIM / 256, 256, 0, stream>>>(sums);
  ra_transpose_values<<<(A_DIM * H_DIM) / 256, 256, 0, stream>>>(values, v_t);
  ra_proj_norm<<<B_DIM / 16, 256, 0, stream>>>(x, W_q, q_bf);
  ra_proj_norm<<<A_DIM / 16, 256, 0, stream>>>(anchors, W_k, k_bf);

  dim3 gsim(A_DIM / 128, B_DIM / 64);
  ra_sim<<<gsim, 256, 0, stream>>>(q_bf, k_bf, sim, sums);

  ra_out<<<B_DIM / 32, 256, 0, stream>>>(sim, v_t, sums, out);
}
